// CausalSelfAttention_8744553415033
// MI455X (gfx1250) — matrix operand order repaired and checked by reading, not compiled, not run on hardware
//
#include <hip/hip_runtime.h>
#include <hip/hip_bf16.h>

// ---------------------------------------------------------------------------
// CDNA5 / gfx1250 causal self-attention:
//   qkv = x @ w_qkv ; flash-attention ; out = att @ w_proj
// Matmuls via V_WMMA_F32_16X16X32_BF16 (wave32, f32 accumulate).
// Global->LDS staging via GLOBAL_LOAD_ASYNC_TO_LDS_B128 (ASYNCcnt),
// double-buffered in the GEMM main loop.
// ---------------------------------------------------------------------------

typedef __attribute__((ext_vector_type(16))) __bf16 v16bf;
typedef __attribute__((ext_vector_type(8)))  float  v8f;

#define WMMA_BF16(a, b, c)                                                   \
  __builtin_amdgcn_wmma_f32_16x16x32_bf16(false, (a), false, (b),            \
                                          (short)0, (c), false, false)

static constexpr int kB = 4;
static constexpr int kT = 2048;
static constexpr int kC = 1024;
static constexpr int kH = 16;
static constexpr int kD = 64;
static constexpr int kMT = kB * kT;      // 8192 rows
static constexpr int kN_QKV = 3 * kC;    // 3072

// --- CDNA5 async global->LDS copy (16B per lane), tracked by ASYNCcnt ------
__device__ __forceinline__ void async_load16(const void* gptr, void* lptr) {
  // LDS aperture maps addr[31:0] directly to the LDS byte offset.
  unsigned lds = (unsigned)(unsigned long long)lptr;
  asm volatile("global_load_async_to_lds_b128 %0, %1, off"
               :: "v"(lds), "v"(gptr)
               : "memory");
}
__device__ __forceinline__ void wait_async0() {
  asm volatile("s_wait_asynccnt 0x0" ::: "memory");
}

// ---------------------------------------------------------------------------
// Prep kernels: fp32 -> bf16 convert / transpose (one-time, bandwidth-trivial)
// ---------------------------------------------------------------------------
__global__ void cvt_f32_to_bf16(const float* __restrict__ in,
                                __bf16* __restrict__ out, int n) {
  int i = blockIdx.x * blockDim.x + threadIdx.x;
  int stride = gridDim.x * blockDim.x;
  for (; i < n; i += stride) out[i] = (__bf16)in[i];
}

// in: [rows][cols] f32  ->  out: [cols][rows] bf16   (N x K layout for WMMA B)
__global__ void transpose_f32_to_bf16(const float* __restrict__ in,
                                      __bf16* __restrict__ out,
                                      int rows, int cols) {
  int i = blockIdx.x * blockDim.x + threadIdx.x;
  int stride = gridDim.x * blockDim.x;
  int n = rows * cols;
  for (; i < n; i += stride) {
    int r = i % rows;   // input row  == output column
    int c = i / rows;   // input col  == output row
    out[i] = (__bf16)in[r * cols + c];
  }
}

// ---------------------------------------------------------------------------
// Tiled bf16 WMMA GEMM: Cacc(f32) = A[M,K] * Bt[N,K]^T
// Block tile 128x128, 8 waves, each wave 32x64 (2x4 WMMA tiles), K-step 32.
// Two-stage LDS ping-pong fed by async global->LDS copies.
// MODE 0: scatter to head-split Q / K / Vt(bf16, V transposed)
// MODE 1: dense f32 store to outf
// ---------------------------------------------------------------------------
template <int MODE>
__global__ __launch_bounds__(256)
void gemm_bf16_wmma(const __bf16* __restrict__ A,
                    const __bf16* __restrict__ Bt,
                    int M, int N, int K,
                    __bf16* __restrict__ qo, __bf16* __restrict__ ko,
                    __bf16* __restrict__ vto, float* __restrict__ outf) {
  constexpr int LDA = 40;  // 32 + 8 pad (keeps 16B alignment, skews banks)
  __shared__ __bf16 As[2][128 * LDA];
  __shared__ __bf16 Bs[2][128 * LDA];

  const int tid  = threadIdx.x;
  const int lane = tid & 31;
  const int wave = tid >> 5;
  const int wm   = wave >> 1;   // 0..3  (M direction)
  const int wn   = wave & 1;    // 0..1  (N direction)
  const int l    = lane & 15;
  const int hi   = lane >> 4;

  const long m0 = (long)blockIdx.y * 128;
  const long n0 = (long)blockIdx.x * 128;

  // staging coordinates: each thread copies 2x16B of A and of B per slab
  const int sr = tid >> 2;           // 0..63
  const int sc = (tid & 3) * 8;      // 0,8,16,24

  auto stage = [&](int buf, int k0) {
#pragma unroll
    for (int p = 0; p < 2; ++p) {
      int rr = sr + p * 64;
      async_load16(A  + (m0 + rr) * (long)K + k0 + sc, &As[buf][rr * LDA + sc]);
      async_load16(Bt + (n0 + rr) * (long)K + k0 + sc, &Bs[buf][rr * LDA + sc]);
    }
  };

  v8f acc[2][4] = {};

  stage(0, 0);
  wait_async0();
  __syncthreads();

  for (int k0 = 0; k0 < K; k0 += 32) {
    const int buf = (k0 >> 5) & 1;
    if (k0 + 32 < K) stage(buf ^ 1, k0 + 32);   // overlap copy with WMMA

    // A fragments: lane<16 holds K 0-7 & 16-23, lane>=16 holds 8-15 & 24-31
    v16bf afrag[2];
#pragma unroll
    for (int mi = 0; mi < 2; ++mi) {
      int row = wm * 32 + mi * 16 + l;
      int c0  = hi * 8;
      union { v16bf v; uint4 u[2]; } ua;
      ua.u[0] = *(const uint4*)(&As[buf][row * LDA + c0]);
      ua.u[1] = *(const uint4*)(&As[buf][row * LDA + c0 + 16]);
      afrag[mi] = ua.v;
    }
    // B fragments: lane<16 holds K 0-7 & 16-23, lane>=16 holds 8-15 & 24-31
#pragma unroll
    for (int ni = 0; ni < 4; ++ni) {
      int nrow = wn * 64 + ni * 16 + l;
      int c0   = hi * 8;
      union { v16bf v; uint4 u[2]; } ub;
      ub.u[0] = *(const uint4*)(&Bs[buf][nrow * LDA + c0]);
      ub.u[1] = *(const uint4*)(&Bs[buf][nrow * LDA + c0 + 16]);
      v16bf bfrag = ub.v;
#pragma unroll
      for (int mi = 0; mi < 2; ++mi)
        acc[mi][ni] = WMMA_BF16(afrag[mi], bfrag, acc[mi][ni]);
    }

    wait_async0();     // our async copies for the next slab have landed
    __syncthreads();   // everyone's reads of `buf` done, next slab visible
  }

  // ---- epilogue: C layout = lane col (lane&15), row = vgpr + 8*(lane>>4) ----
#pragma unroll
  for (int mi = 0; mi < 2; ++mi) {
#pragma unroll
    for (int ni = 0; ni < 4; ++ni) {
      long n = n0 + wn * 64 + ni * 16 + l;
      if (MODE == 0) {
        int region = (int)(n >> 10);      // 0=Q 1=K 2=V
        int cc = (int)(n & 1023);
        int h  = cc >> 6;
        int d  = cc & 63;
#pragma unroll
        for (int r = 0; r < 8; ++r) {
          long mrow = m0 + wm * 32 + mi * 16 + hi * 8 + r;
          int b = (int)(mrow >> 11);      // / T
          int t = (int)(mrow & 2047);
          long bh = (long)b * kH + h;
          __bf16 val = (__bf16)acc[mi][ni][r];
          if (region == 0)      qo [(bh * kT + t) * kD + d] = val;
          else if (region == 1) ko [(bh * kT + t) * kD + d] = val;
          else                  vto[(bh * kD + d) * kT + t] = val;  // V^T
        }
      } else {
#pragma unroll
        for (int r = 0; r < 8; ++r) {
          long mrow = m0 + wm * 32 + mi * 16 + hi * 8 + r;
          outf[mrow * (long)N + n] = acc[mi][ni][r];
        }
      }
    }
  }
}

// ---------------------------------------------------------------------------
// Flash attention (causal, online softmax), one block = 128 q-rows x one (b,h)
// 8 waves; each wave owns 16 q-rows. Per k-tile of 128:
//   S(16x128) = Q * K^T  -> 16 WMMAs ; O(16x64) += P * V -> 16 WMMAs
// K/V tiles staged with async global->LDS copies.
// ---------------------------------------------------------------------------
__global__ __launch_bounds__(256)
void flash_attn_wmma(const __bf16* __restrict__ Q,
                     const __bf16* __restrict__ Kg,
                     const __bf16* __restrict__ Vt,
                     __bf16* __restrict__ attO) {
  constexpr int LKS = 72;    // K tile stride  (64 + 8)
  constexpr int LVS = 136;   // V^T tile stride (128 + 8)
  constexpr int LPS = 72;    // P scratch stride (64 + 8)
  __shared__ __bf16 Ks[128 * LKS];      // K tile  [t_k][d]
  __shared__ __bf16 Vs[64  * LVS];      // V^T tile [d][t_k]
  __shared__ __bf16 Ps[8 * 16 * LPS];   // per-wave P re-layout scratch

  const int tid  = threadIdx.x;
  const int lane = tid & 31;
  const int wave = tid >> 5;
  const int l    = lane & 15;
  const int hi   = lane >> 4;
  const int qi   = blockIdx.x;          // q tile index (128 rows)
  const int bh   = blockIdx.y;          // b*H + h
  const int b    = bh >> 4;
  const int h    = bh & 15;

  const __bf16* Qbh = Q  + (long)bh * kT * kD;
  const __bf16* Kbh = Kg + (long)bh * kT * kD;
  const __bf16* Vbh = Vt + (long)bh * kD * kT;

  // Q fragments for this wave's 16 rows (A-matrix 16x64 => 2 frags), direct
  // from global: row = base + (lane&15); cols per the 16-bit A layout.
  v16bf qfrag[2];
  {
    const __bf16* qp = Qbh + (long)(qi * 128 + wave * 16 + l) * kD;
#pragma unroll
    for (int kb = 0; kb < 2; ++kb) {
      int c0 = kb * 32 + hi * 8;
      union { v16bf v; uint4 u[2]; } ua;
      ua.u[0] = *(const uint4*)(qp + c0);
      ua.u[1] = *(const uint4*)(qp + c0 + 16);
      qfrag[kb] = ua.v;
    }
  }

  v8f o[4] = {};
  float mrow[8], lrow[8];
#pragma unroll
  for (int r = 0; r < 8; ++r) { mrow[r] = -3.0e38f; lrow[r] = 0.0f; }
  const float scale = 0.125f;  // 1/sqrt(64)

  for (int j = 0; j <= qi; ++j) {       // causal: only k-tiles <= q-tile
    __syncthreads();
    {  // async-stage K tile 128x64 and V^T tile 64x128 into LDS
      int r  = tid >> 3;
      int cc = (tid & 7) * 8;
#pragma unroll
      for (int p = 0; p < 4; ++p) {
        int rr = r + p * 32;
        async_load16(Kbh + (long)(j * 128 + rr) * kD + cc, &Ks[rr * LKS + cc]);
      }
      int r2 = tid >> 4;
      int c2 = (tid & 15) * 8;
#pragma unroll
      for (int p = 0; p < 4; ++p) {
        int rr = r2 + p * 16;
        async_load16(Vbh + (long)rr * kT + j * 128 + c2, &Vs[rr * LVS + c2]);
      }
    }
    wait_async0();
    __syncthreads();

    // S = Q K^T : 8 column tiles x 2 K-steps
    v8f s[8] = {};
#pragma unroll
    for (int j2 = 0; j2 < 8; ++j2) {
#pragma unroll
      for (int kb = 0; kb < 2; ++kb) {
        int nrow = j2 * 16 + l;
        int c0   = kb * 32 + hi * 8;
        union { v16bf v; uint4 u[2]; } ub;
        ub.u[0] = *(const uint4*)(&Ks[nrow * LKS + c0]);
        ub.u[1] = *(const uint4*)(&Ks[nrow * LKS + c0 + 16]);
        s[j2] = WMMA_BF16(qfrag[kb], ub.v, s[j2]);
      }
    }

    // scale + causal mask (lane holds column k = j*128 + j2*16 + l)
#pragma unroll
    for (int j2 = 0; j2 < 8; ++j2) {
      int kcol = j * 128 + j2 * 16 + l;
#pragma unroll
      for (int r = 0; r < 8; ++r) {
        int qrow = qi * 128 + wave * 16 + hi * 8 + r;
        float v = s[j2][r] * scale;
        s[j2][r] = (kcol <= qrow) ? v : -3.0e38f;
      }
    }

    // online softmax: row max across 8 tiles + 16-lane shuffle reduction
    float mnew[8];
#pragma unroll
    for (int r = 0; r < 8; ++r) {
      float mx = mrow[r];
#pragma unroll
      for (int j2 = 0; j2 < 8; ++j2) mx = fmaxf(mx, s[j2][r]);
      mx = fmaxf(mx, __shfl_xor(mx, 8, 32));
      mx = fmaxf(mx, __shfl_xor(mx, 4, 32));
      mx = fmaxf(mx, __shfl_xor(mx, 2, 32));
      mx = fmaxf(mx, __shfl_xor(mx, 1, 32));
      mnew[r] = mx;
    }
#pragma unroll
    for (int r = 0; r < 8; ++r) {
      float sc = __expf(mrow[r] - mnew[r]);
      lrow[r] *= sc;
#pragma unroll
      for (int nt = 0; nt < 4; ++nt) o[nt][r] *= sc;
      mrow[r] = mnew[r];
    }

    // P = exp(S - m); re-layout C->A through wave-private LDS, O += P * V
    __bf16* Pw = &Ps[wave * 16 * LPS];
#pragma unroll
    for (int half = 0; half < 2; ++half) {
#pragma unroll
      for (int j2 = 0; j2 < 4; ++j2) {
        int sj = half * 4 + j2;
#pragma unroll
        for (int r = 0; r < 8; ++r) {
          float p = __expf(s[sj][r] - mrow[r]);
          lrow[r] += p;   // lane-partial row sum (reduced at the end)
          Pw[(hi * 8 + r) * LPS + j2 * 16 + l] = (__bf16)p;
        }
      }
#pragma unroll
      for (int kb = 0; kb < 2; ++kb) {
        int c0 = kb * 32 + hi * 8;
        union { v16bf v; uint4 u[2]; } up;
        up.u[0] = *(const uint4*)(&Pw[l * LPS + c0]);
        up.u[1] = *(const uint4*)(&Pw[l * LPS + c0 + 16]);
        v16bf pfrag = up.v;
        int kglob = half * 2 + kb;
#pragma unroll
        for (int nt = 0; nt < 4; ++nt) {
          int nrow = nt * 16 + l;
          int c1   = kglob * 32 + hi * 8;
          union { v16bf v; uint4 u[2]; } uv;
          uv.u[0] = *(const uint4*)(&Vs[nrow * LVS + c1]);
          uv.u[1] = *(const uint4*)(&Vs[nrow * LVS + c1 + 16]);
          o[nt] = WMMA_BF16(pfrag, uv.v, o[nt]);
        }
      }
    }
  }

  // finalize: reduce row sums across lanes, normalize, store merged heads
#pragma unroll
  for (int r = 0; r < 8; ++r) {
    float ls = lrow[r];
    ls += __shfl_xor(ls, 8, 32);
    ls += __shfl_xor(ls, 4, 32);
    ls += __shfl_xor(ls, 2, 32);
    ls += __shfl_xor(ls, 1, 32);
    float inv = 1.0f / ls;
    int  qrow = qi * 128 + wave * 16 + hi * 8 + r;
    long orow = (long)b * kT + qrow;
#pragma unroll
    for (int nt = 0; nt < 4; ++nt) {
      int col = h * 64 + nt * 16 + l;
      attO[orow * kC + col] = (__bf16)(o[nt][r] * inv);
    }
  }
}

// ---------------------------------------------------------------------------
// Host launcher
// ---------------------------------------------------------------------------
extern "C" void kernel_launch(void* const* d_in, const int* in_sizes, int n_in,
                              void* d_out, int out_size, void* d_ws,
                              size_t ws_size, hipStream_t stream) {
  const float* x      = (const float*)d_in[0];
  const float* w_qkv  = (const float*)d_in[1];
  const float* w_proj = (const float*)d_in[2];
  float* out = (float*)d_out;

  char* ws = (char*)d_ws;
  size_t off = 0;
  auto alloc = [&](size_t bytes) -> void* {
    void* p = ws + off;
    off += (bytes + 255) & ~(size_t)255;
    return p;
  };
  __bf16* xb     = (__bf16*)alloc((size_t)kMT * kC * 2);        // 16 MB
  __bf16* wqkvT  = (__bf16*)alloc((size_t)kN_QKV * kC * 2);     //  6 MB
  __bf16* wprojT = (__bf16*)alloc((size_t)kC * kC * 2);         //  2 MB
  __bf16* Qb     = (__bf16*)alloc((size_t)kMT * kC * 2);        // 16 MB
  __bf16* Kb     = (__bf16*)alloc((size_t)kMT * kC * 2);        // 16 MB
  __bf16* Vtb    = (__bf16*)alloc((size_t)kMT * kC * 2);        // 16 MB
  __bf16* attO   = (__bf16*)alloc((size_t)kMT * kC * 2);        // 16 MB

  // 1) precision/layout prep
  cvt_f32_to_bf16<<<4096, 256, 0, stream>>>(x, xb, kMT * kC);
  transpose_f32_to_bf16<<<2048, 256, 0, stream>>>(w_qkv, wqkvT, kC, kN_QKV);
  transpose_f32_to_bf16<<<1024, 256, 0, stream>>>(w_proj, wprojT, kC, kC);

  // 2) fused QKV GEMM -> head-split Q, K, V^T (bf16)
  gemm_bf16_wmma<0><<<dim3(kN_QKV / 128, kMT / 128), 256, 0, stream>>>(
      xb, wqkvT, kMT, kN_QKV, kC, Qb, Kb, Vtb, nullptr);

  // 3) causal flash attention
  flash_attn_wmma<<<dim3(kT / 128, kB * kH), 256, 0, stream>>>(Qb, Kb, Vtb,
                                                               attO);

  // 4) output projection -> f32
  gemm_bf16_wmma<1><<<dim3(kC / 128, kMT / 128), 256, 0, stream>>>(
      attO, wprojT, kMT, kC, kC, nullptr, nullptr, nullptr, out);
}
